// DetailCapture_33629593928047
// MI455X (gfx1250) — matrix operand order repaired and checked by reading, not compiled, not run on hardware
//
#include <hip/hip_runtime.h>
#include <math.h>

// ---------------------------------------------------------------------------
// Problem constants (fixed by the reference: B=2, C=256, H=W=128)
// ---------------------------------------------------------------------------
#define B_   2
#define C_   256
#define H_   128
#define W_   128
#define HW_  (H_ * W_)          // 16384
#define M_   (B_ * HW_)         // 32768 pixels
#define KOFF 2304               // 9 * 256 im2col depth for offset conv
#define EPS_ 1e-6f

typedef unsigned short u16;
typedef __attribute__((ext_vector_type(16))) __bf16 v16bf;
typedef __attribute__((ext_vector_type(8)))  float  v8f;

// ---------------------------------------------------------------------------
// Small helpers
// ---------------------------------------------------------------------------
__device__ __forceinline__ u16 f2bf(float f) {
  unsigned int u = __float_as_uint(f);
  u += 0x7fffu + ((u >> 16) & 1u);         // round-to-nearest-even
  return (u16)(u >> 16);
}
__device__ __forceinline__ float gelu_exact(float v) {
  return 0.5f * v * (1.0f + erff(v * 0.70710678118654752f));
}

union FragU { v16bf v; uint4 q[2]; };

__device__ __forceinline__ v16bf load_frag2(const u16* p0, const u16* p1) {
  FragU f;
  f.q[0] = *(const uint4*)(const void*)p0;
  f.q[1] = *(const uint4*)(const void*)p1;
  return f.v;
}
__device__ __forceinline__ v16bf zero_frag() {
  FragU f;
  f.q[0] = make_uint4(0u, 0u, 0u, 0u);
  f.q[1] = make_uint4(0u, 0u, 0u, 0u);
  return f.v;
}
__device__ __forceinline__ v8f zero8() {
  v8f z;
  for (int i = 0; i < 8; ++i) z[i] = 0.0f;
  return z;
}

// CDNA5 async Global->LDS copy (16B), tracked by ASYNCcnt.
__device__ __forceinline__ void async_copy_b128(unsigned lds_off,
                                                const void* gptr) {
  unsigned long long ga = (unsigned long long)gptr;
  asm volatile("global_load_async_to_lds_b128 %0, %1, off"
               :: "v"(lds_off), "v"(ga) : "memory");
}
__device__ __forceinline__ void wait_asynccnt0() {
  asm volatile("s_wait_asynccnt 0x0" ::: "memory");
}

__device__ __forceinline__ float block_sum_256(float v, float* sdata) {
  int t = threadIdx.x;
  sdata[t] = v;
  __syncthreads();
  for (int s = 128; s > 0; s >>= 1) {
    if (t < s) sdata[t] += sdata[t + s];
    __syncthreads();
  }
  float r = sdata[0];
  __syncthreads();
  return r;
}

// ---------------------------------------------------------------------------
// K0: NCHW fp32 -> NHWC fp32 (for gathers) + NHWC bf16 (WMMA A operand)
// ---------------------------------------------------------------------------
__global__ void to_nhwc_kernel(const float* __restrict__ x,
                               float* __restrict__ xn,
                               u16* __restrict__ xh) {
  const size_t total = (size_t)M_ * C_;
  for (size_t o = (size_t)blockIdx.x * blockDim.x + threadIdx.x; o < total;
       o += (size_t)gridDim.x * blockDim.x) {
    int c   = (int)(o & (C_ - 1));
    size_t p = o >> 8;                       // pixel index
    int b   = (int)(p >> 14);
    int hw  = (int)(p & (HW_ - 1));
    float v = x[(((size_t)b * C_ + c) << 14) + hw];
    xn[o] = v;
    xh[o] = f2bf(v);
  }
}

// ---------------------------------------------------------------------------
// K1a: pack offset-conv weights [18,256,3,3] -> bf16 [32(pad)][K=2304],
//      K index = k9*256 + c (matches A im2col ordering), 3 branches.
// ---------------------------------------------------------------------------
__global__ void pack_offw_kernel(const float* __restrict__ w0,
                                 const float* __restrict__ w1,
                                 const float* __restrict__ w2,
                                 u16* __restrict__ wp) {
  const int total = 3 * 32 * KOFF;
  for (int idx = blockIdx.x * blockDim.x + threadIdx.x; idx < total;
       idx += gridDim.x * blockDim.x) {
    int br = idx / (32 * KOFF);
    int r  = idx - br * (32 * KOFF);
    int n  = r / KOFF;
    int kk = r - n * KOFF;
    int k9 = kk >> 8;                        // 0..8 kernel point
    int cc = kk & (C_ - 1);                  // input channel
    float v = 0.0f;
    if (n < 18) {
      const float* w = (br == 0) ? w0 : (br == 1) ? w1 : w2;
      v = w[((size_t)n * C_ + cc) * 9 + k9]; // OIHW flat
    }
    wp[idx] = f2bf(v);
  }
}

// K1b: plain fp32 -> bf16 copy (conv_w / mlp_w1 / mlp_w2, already [N][K])
__global__ void f2bf_copy_kernel(const float* __restrict__ src,
                                 u16* __restrict__ dst, int n) {
  for (int i = blockIdx.x * blockDim.x + threadIdx.x; i < n;
       i += gridDim.x * blockDim.x) dst[i] = f2bf(src[i]);
}

// ---------------------------------------------------------------------------
// K2: dense WMMA GEMM  out[M][N] = A[M][K](bf16) * B[N][K](bf16)^T + bias.
//     One block (4 waves) computes a 64x64 tile. The shared 64xK bf16 B
//     panel is staged once into LDS with async Global->LDS copies
//     (ASYNCcnt + s_wait_asynccnt), then WMMA B fragments come from LDS
//     (ds_load_b128) while the streaming A operand stays on the VMEM pipe.
//     MODE 0: fp32 out, MODE 1: GELU -> bf16 out.
// ---------------------------------------------------------------------------
template <int MODE>
__global__ void gemm_kernel(const u16* __restrict__ A, int lda,
                            const u16* __restrict__ Bw, int ldb,
                            const float* __restrict__ bias,
                            float* __restrict__ outF, u16* __restrict__ outH,
                            int M, int N, int K) {
  extern __shared__ u16 bpanel[];            // [64][K] bf16
  const int wave = threadIdx.x >> 5;
  const int lane = threadIdx.x & 31;
  const int tilesN = N >> 6;
  const int tileM = blockIdx.x / tilesN;
  const int tileN = blockIdx.x - tileM * tilesN;
  const int nbase = tileN << 6;

  // ---- async-stage the 64 x K B panel into LDS ----
  const unsigned lds0 = (unsigned)(size_t)(void*)bpanel;
  const int chunks = (K << 6) >> 3;          // 64*K bf16 / 8 per 16B chunk
  for (int idx = threadIdx.x; idx < chunks; idx += blockDim.x) {
    const int row = (idx << 3) / K;
    const int col = (idx << 3) - row * K;
    async_copy_b128(lds0 + ((unsigned)idx << 4),
                    Bw + (size_t)(nbase + row) * ldb + col);
  }
  wait_asynccnt0();
  __syncthreads();

  const int mbase = (tileM << 6) + (wave << 4);
  if (mbase >= M) return;

  const u16* arow = A + (size_t)(mbase + (lane & 15)) * lda;
  const int asel = (lane & 16) ? 8 : 0;      // 16-bit A 16x32 lane layout
  const int bsel = (lane & 16) ? 8 : 0;      // 16-bit B 32x16 lane layout

  v8f acc[4];
#pragma unroll
  for (int j = 0; j < 4; ++j) acc[j] = zero8();

  for (int kb = 0; kb < K; kb += 32) {
    __builtin_prefetch(arow + kb + 128, 0, 0);   // global_prefetch_b8
    v16bf a = load_frag2(arow + kb + asel, arow + kb + 16 + asel);
#pragma unroll
    for (int j = 0; j < 4; ++j) {
      const u16* brow = bpanel + (size_t)(j * 16 + (lane & 15)) * K + kb + bsel;
      v16bf b = load_frag2(brow, brow + 16);   // ds_load_b128 x2
      acc[j] = __builtin_amdgcn_wmma_f32_16x16x32_bf16(
          false, a, false, b, (short)0, acc[j], false, false);
    }
  }

  const int mrow0 = mbase + ((lane & 16) ? 8 : 0);
#pragma unroll
  for (int j = 0; j < 4; ++j) {
    int n = nbase + j * 16 + (lane & 15);
    float bv = bias ? bias[n] : 0.0f;
#pragma unroll
    for (int r = 0; r < 8; ++r) {
      float v = acc[j][r] + bv;
      size_t o = (size_t)(mrow0 + r) * N + n;
      if (MODE == 1) {
        outH[o] = f2bf(gelu_exact(v));
      } else {
        outF[o] = v;
      }
    }
  }
}

// ---------------------------------------------------------------------------
// K3: offset conv as implicit-im2col WMMA GEMM.
//     A: 16 pixels x 32 channels from shifted NHWC location (zero at pad),
//     B: packed weights [32][2304] (L2-resident). Writes off[pixel][18] fp32.
// ---------------------------------------------------------------------------
__global__ void offset_gemm_kernel(const u16* __restrict__ xh,
                                   const u16* __restrict__ wp,
                                   float* __restrict__ off,
                                   int pad, int dil) {
  const int wave = threadIdx.x >> 5;
  const int lane = threadIdx.x & 31;
  const int wt = blockIdx.x * (blockDim.x >> 5) + wave;
  if (wt >= (M_ >> 4)) return;
  const int mbase = wt << 4;
  const int m = mbase + (lane & 15);
  const int b = m >> 14;
  const int rem = m & (HW_ - 1);
  const int y = rem >> 7;
  const int x = rem & (W_ - 1);
  const int asel = (lane & 16) ? 8 : 0;
  const int bsel = (lane & 16) ? 8 : 0;
  const int n0 = lane & 15;

  v8f a0 = zero8(), a1 = zero8();

  for (int k9 = 0; k9 < 9; ++k9) {
    const int ky = k9 / 3, kx = k9 - ky * 3;
    const int sy = y - pad + ky * dil;
    const int sx = x - pad + kx * dil;
    const bool valid = (sy >= 0) && (sy < H_) && (sx >= 0) && (sx < W_);
    const int cy = valid ? sy : 0, cx = valid ? sx : 0;
    const u16* arow = xh + ((size_t)((b * H_ + cy) * W_ + cx)) * C_;
    const int kb9 = k9 << 8;
#pragma unroll
    for (int cb = 0; cb < C_; cb += 32) {
      v16bf a = valid ? load_frag2(arow + cb + asel, arow + cb + 16 + asel)
                      : zero_frag();
      const u16* b0 = wp + (size_t)n0 * KOFF + kb9 + cb + bsel;
      const u16* b1 = wp + (size_t)(16 + n0) * KOFF + kb9 + cb + bsel;
      a0 = __builtin_amdgcn_wmma_f32_16x16x32_bf16(
          false, a, false, load_frag2(b0, b0 + 16), (short)0, a0, false, false);
      a1 = __builtin_amdgcn_wmma_f32_16x16x32_bf16(
          false, a, false, load_frag2(b1, b1 + 16), (short)0, a1, false, false);
    }
  }

  const int mrow0 = mbase + ((lane & 16) ? 8 : 0);
#pragma unroll
  for (int r = 0; r < 8; ++r)            // n0 in [0,16) < 18: always valid
    off[(size_t)(mrow0 + r) * 18 + n0] = a0[r];
  if (n0 + 16 < 18) {
#pragma unroll
    for (int r = 0; r < 8; ++r)
      off[(size_t)(mrow0 + r) * 18 + n0 + 16] = a1[r];
  }
}

// ---------------------------------------------------------------------------
// K4: fused deformable depthwise 3x3 sample + LayerNorm2d + GELU,
//     accumulated into the running sum s (already holds the 1x1-conv path).
//     One block = one pixel, 256 threads = 256 channels (gathers coalesce
//     along the NHWC channel dim; bilinear coords uniform per block, so the
//     L2-resident x tile is streamed with unit-stride b32 loads).
// ---------------------------------------------------------------------------
__global__ void deform_branch_kernel(const float* __restrict__ xn,
                                     const float* __restrict__ off,
                                     const float* __restrict__ defw,
                                     const float* __restrict__ lw,
                                     const float* __restrict__ lb,
                                     float* __restrict__ s,
                                     int pad, int dil) {
  __shared__ float soff[18];
  __shared__ float sdata[256];
  const int c = threadIdx.x;
  const size_t p = blockIdx.x;
  const int b = (int)(p >> 14);
  const int y = (int)((p >> 7) & (H_ - 1));
  const int x = (int)(p & (W_ - 1));
  if (c < 18) soff[c] = off[p * 18 + c];
  __syncthreads();

  float acc = 0.0f;
#pragma unroll
  for (int k = 0; k < 9; ++k) {
    const int ky = k / 3, kx = k - ky * 3;
    const float py = (float)(y - pad + ky * dil) + soff[2 * k + 0];
    const float px = (float)(x - pad + kx * dil) + soff[2 * k + 1];
    const float y0 = floorf(py), x0 = floorf(px);
    float val = 0.0f;
#pragma unroll
    for (int dy = 0; dy < 2; ++dy) {
#pragma unroll
      for (int dx = 0; dx < 2; ++dx) {
        const float yc = y0 + (float)dy, xc = x0 + (float)dx;
        const bool valid = (yc >= 0.0f) && (yc < (float)H_) &&
                           (xc >= 0.0f) && (xc < (float)W_);
        if (valid) {
          const float w = (1.0f - fabsf(py - yc)) * (1.0f - fabsf(px - xc));
          const int yi = (int)yc, xi = (int)xc;
          val += w * xn[((size_t)((b * H_ + yi) * W_ + xi)) * C_ + c];
        }
      }
    }
    acc += defw[c * 9 + k] * val;
  }

  // LayerNorm2d over channels (block == channel dim), then exact GELU.
  const float mean = block_sum_256(acc, sdata) * (1.0f / C_);
  const float d = acc - mean;
  const float var = block_sum_256(d * d, sdata) * (1.0f / C_);
  const float yv = lw[c] * (d * rsqrtf(var + EPS_)) + lb[c];
  s[p * C_ + c] += gelu_exact(yv);
}

// ---------------------------------------------------------------------------
// K5: LayerNorm2d (optionally +residual). Emits fp32, bf16 and/or NCHW views.
// ---------------------------------------------------------------------------
__global__ void ln_kernel(const float* __restrict__ in,
                          const float* __restrict__ res,
                          const float* __restrict__ w,
                          const float* __restrict__ bb,
                          float* __restrict__ outF,
                          u16* __restrict__ outH,
                          float* __restrict__ outNCHW) {
  __shared__ float sdata[256];
  const int c = threadIdx.x;
  const size_t p = blockIdx.x;
  float v = in[p * C_ + c];
  if (res) v += res[p * C_ + c];
  const float mean = block_sum_256(v, sdata) * (1.0f / C_);
  const float d = v - mean;
  const float var = block_sum_256(d * d, sdata) * (1.0f / C_);
  const float yv = w[c] * (d * rsqrtf(var + EPS_)) + bb[c];
  if (outF) outF[p * C_ + c] = yv;
  if (outH) outH[p * C_ + c] = f2bf(yv);
  if (outNCHW) {
    const int b = (int)(p >> 14);
    const int hw = (int)(p & (HW_ - 1));
    outNCHW[(((size_t)b * C_ + c) << 14) + hw] = yv;
  }
}

// ---------------------------------------------------------------------------
// Host side
// ---------------------------------------------------------------------------
extern "C" void kernel_launch(void* const* d_in, const int* in_sizes, int n_in,
                              void* d_out, int out_size, void* d_ws,
                              size_t ws_size, hipStream_t stream) {
  const float* x      = (const float*)d_in[0];
  const float* off_w[3] = {(const float*)d_in[1], (const float*)d_in[5],
                           (const float*)d_in[9]};
  const float* def_w[3] = {(const float*)d_in[2], (const float*)d_in[6],
                           (const float*)d_in[10]};
  const float* bw[3]    = {(const float*)d_in[3], (const float*)d_in[7],
                           (const float*)d_in[11]};
  const float* bb[3]    = {(const float*)d_in[4], (const float*)d_in[8],
                           (const float*)d_in[12]};
  const float* conv_w = (const float*)d_in[13];
  const float* conv_b = (const float*)d_in[14];
  const float* n1_w   = (const float*)d_in[15];
  const float* n1_b   = (const float*)d_in[16];
  const float* n2_w   = (const float*)d_in[17];
  const float* n2_b   = (const float*)d_in[18];
  const float* mlp_w1 = (const float*)d_in[19];
  const float* mlp_b1 = (const float*)d_in[20];
  const float* mlp_w2 = (const float*)d_in[21];
  const float* mlp_b2 = (const float*)d_in[22];

  // Workspace carve (256-byte aligned slabs).
  char* wp_ = (char*)d_ws;
  auto carve = [&](size_t bytes) -> void* {
    void* p = (void*)wp_;
    wp_ += (bytes + 255) & ~(size_t)255;
    return p;
  };
  float* xn   = (float*)carve((size_t)M_ * C_ * 4);        // NHWC fp32
  u16*   xh   = (u16*)  carve((size_t)M_ * C_ * 2);        // NHWC bf16
  u16*   wpk  = (u16*)  carve((size_t)3 * 32 * KOFF * 2);  // packed offset W
  u16*   cwh  = (u16*)  carve((size_t)C_ * C_ * 2);        // conv_w bf16
  u16*   w1h  = (u16*)  carve((size_t)512 * 256 * 2);      // mlp_w1 bf16
  u16*   w2h  = (u16*)  carve((size_t)256 * 512 * 2);      // mlp_w2 bf16
  float* offb = (float*)carve((size_t)M_ * 18 * 4);        // offsets (reused)
  float* ssum = (float*)carve((size_t)M_ * C_ * 4);        // x1+x2+x3+c
  float* out1 = (float*)carve((size_t)M_ * C_ * 4);        // LN1 (residual)
  u16*   o1h  = (u16*)  carve((size_t)M_ * C_ * 2);        // LN1 bf16
  u16*   hbuf = (u16*)  carve((size_t)M_ * 512 * 2);       // MLP hidden bf16
  float* tbuf = (float*)carve((size_t)M_ * C_ * 4);        // MLP out fp32

  // Layout transforms + weight packing.
  to_nhwc_kernel<<<4096, 256, 0, stream>>>(x, xn, xh);
  pack_offw_kernel<<<864, 256, 0, stream>>>(off_w[0], off_w[1], off_w[2], wpk);
  f2bf_copy_kernel<<<256, 256, 0, stream>>>(conv_w, cwh, C_ * C_);
  f2bf_copy_kernel<<<512, 256, 0, stream>>>(mlp_w1, w1h, 512 * 256);
  f2bf_copy_kernel<<<512, 256, 0, stream>>>(mlp_w2, w2h, 256 * 512);

  // 1x1 conv initializes the sum buffer: 64x64 tiles -> (512 M)*(4 N) blocks.
  gemm_kernel<0><<<2048, 128, 64 * 256 * 2, stream>>>(
      xh, C_, cwh, C_, conv_b, ssum, nullptr, M_, C_, C_);

  // Deformable branches (pad == dil: 1, 9, 12), each accumulates into ssum.
  const int pd[3] = {1, 9, 12};
  for (int i = 0; i < 3; ++i) {
    offset_gemm_kernel<<<512, 128, 0, stream>>>(xh, wpk + (size_t)i * 32 * KOFF,
                                                offb, pd[i], pd[i]);
    deform_branch_kernel<<<M_, 256, 0, stream>>>(xn, offb, def_w[i], bw[i],
                                                 bb[i], ssum, pd[i], pd[i]);
  }

  // LN1 -> out1 (fp32 residual) + bf16 for MLP GEMM.
  ln_kernel<<<M_, 256, 0, stream>>>(ssum, nullptr, n1_w, n1_b, out1, o1h,
                                    nullptr);

  // MLP: (M,512,K=256) with GELU->bf16, then (M,256,K=512) fp32.
  gemm_kernel<1><<<4096, 128, 64 * 256 * 2, stream>>>(
      o1h, 256, w1h, 256, mlp_b1, nullptr, hbuf, M_, 512, 256);
  gemm_kernel<0><<<2048, 128, 64 * 512 * 2, stream>>>(
      hbuf, 512, w2h, 512, mlp_b2, tbuf, nullptr, M_, 256, 512);

  // Residual + LN2, store transposed back to NCHW fp32 output.
  ln_kernel<<<M_, 256, 0, stream>>>(tbuf, out1, n2_w, n2_b, nullptr, nullptr,
                                    (float*)d_out);
}
